// GRUGANGenerator_8641474200351
// MI455X (gfx1250) — matrix operand order repaired and checked by reading, not compiled, not run on hardware
//
#include <hip/hip_runtime.h>
#include <hip/hip_bf16.h>

typedef __attribute__((ext_vector_type(16))) _Float16 v16h;
typedef __attribute__((ext_vector_type(8)))  float    v8f;

#define BATCH 512
#define HDIM  64
#define FDIM  32
#define SEQ   2048

// ---------------- helpers ----------------

__device__ __forceinline__ v8f wmma_f16(v16h a, v16h b, v8f c) {
  return __builtin_amdgcn_wmma_f32_16x16x32_f16(
      /*neg_a=*/false, a, /*neg_b=*/false, b,
      /*c_mod=*/(short)0, c, /*reuse_a=*/false, /*reuse_b=*/false);
}

// Hardware v_rcp_f32 (avoids the IEEE div-scale + Newton-Raphson chain).
__device__ __forceinline__ float fast_rcp(float x) {
  return __builtin_amdgcn_rcpf(x);
}
__device__ __forceinline__ float sigm_f(float x) {
  return fast_rcp(1.0f + __expf(-x));          // v_exp_f32 + v_rcp_f32
}
__device__ __forceinline__ float tanh_f(float x) {
  return 1.0f - 2.0f * fast_rcp(__expf(2.0f * x) + 1.0f);
}
__device__ __forceinline__ float leaky_f(float x) {
  return x >= 0.0f ? x : 0.01f * x;
}

// Fragment-linear load: each lane reads its 16 contiguous f16 (2x ds_load_b128).
// Used for both A fragments (16x32, state buffers) and B fragments (32x16, weights).
__device__ __forceinline__ v16h load_frag(const _Float16* frag) {
  const int lane = threadIdx.x & 31;
  return *(const v16h*)(frag + lane * 16);
}

// A-fragment element mapping (16x32 f16): lane l holds m = l&15, kh = l>>4;
// slot 2v+j holds k = kh*8 + (v>>2)*16 + (v&3)*2 + j.
// Inverse (for producers): given (m, k):
//   kh = (k>>3)&1 ; v = ((k>>4)<<2) + ((k>>1)&3) ; j = k&1
//   lane' = (m&15) + (kh<<4) ; index = lane'*16 + 2v + j

// Pack global f32 weight W (N x K row-major), used as B^T (K x N) in D = A*B,
// into lane-linear f16 fragments: dst[((nt*KT+kt)*32 + lane)*16 + slot]
// holds W[nt*16 + (lane&15)][kt*32 + (slot>>3)*16 + (lane>>4)*8 + (slot&7)].
__device__ void pack_W(const float* __restrict__ Wg, _Float16* dst, int N, int K) {
  const int KT = K >> 5;
  const int total = (N >> 4) * KT * 512;
  for (int i = threadIdx.x; i < total; i += blockDim.x) {
    const int slot = i & 15;
    const int lane = (i >> 4) & 31;
    const int f    = i >> 9;
    const int kt   = f % KT;
    const int nt   = f / KT;
    const int k = (kt << 5) + ((slot >> 3) << 4) + ((lane >> 4) << 3) + (slot & 7);
    const int n = (nt << 4) + (lane & 15);
    dst[i] = (_Float16)Wg[n * K + k];
  }
}

// ---------------- kernel ----------------
// grid = B/16 blocks (one 16-row batch tile), block = 128 threads = 4 waves.
// Wave w owns H-columns [w*16, w*16+16) of each direction's hidden state,
// lat columns [w*16, w*16+16), and (waves 0,1) output columns [w*16, w*16+16).
// State buffers (x, concat(hf,hb), seq) are stored in LDS as fragment-linear
// 16x32 A-chunks: chunk c covers K-columns [c*32, c*32+32), 512 f16 per chunk.

__global__ void __launch_bounds__(128)
gru_gan_scan_kernel(const float* __restrict__ noise,
                    const float* __restrict__ W_ih_f, const float* __restrict__ W_hh_f,
                    const float* __restrict__ b_ih_f, const float* __restrict__ b_hh_f,
                    const float* __restrict__ W_ih_b, const float* __restrict__ W_hh_b,
                    const float* __restrict__ b_ih_b, const float* __restrict__ b_hh_b,
                    const float* __restrict__ W_lat,  const float* __restrict__ b_lat,
                    const float* __restrict__ W_out,  const float* __restrict__ b_out,
                    float* __restrict__ out) {
  extern __shared__ _Float16 smem[];
  _Float16* wihf = smem;              // 12 nt * 2 kt * 512 = 12288
  _Float16* whhf = wihf + 12288;
  _Float16* wihb = whhf + 12288;
  _Float16* whhb = wihb + 12288;
  _Float16* wlat = whhb + 12288;      // 4 nt * 4 kt * 512 = 8192
  _Float16* wout = wlat + 8192;       // 2 nt * 2 kt * 512 = 2048
  _Float16* hcat = wout + 2048;       // 2 parities * 4 chunks * 512 = 4096
  _Float16* xbuf = hcat + 4096;       // 2 parities * 2 chunks * 512 = 2048
  _Float16* seqb = xbuf + 2048;       // 2 chunks * 512 = 1024

  const int lane = threadIdx.x & 31;
  const int w    = threadIdx.x >> 5;           // wave id 0..3
  const int rowbase = blockIdx.x * 16;

  // Warm L2 with the weights before packing.
  __builtin_prefetch(W_ih_f, 0, 1);
  __builtin_prefetch(W_hh_f, 0, 1);
  __builtin_prefetch(W_ih_b, 0, 1);
  __builtin_prefetch(W_hh_b, 0, 1);
  __builtin_prefetch(W_lat,  0, 1);

  // One-time fragment packing (f32 -> f16, lane-linear B fragments).
  pack_W(W_ih_f, wihf, 192, 64);
  pack_W(W_hh_f, whhf, 192, 64);
  pack_W(W_ih_b, wihb, 192, 64);
  pack_W(W_hh_b, whhb, 192, 64);
  pack_W(W_lat,  wlat,  64, 128);
  pack_W(W_out,  wout,  32, 64);

  // Initial state (parity 0): x = 0, h_f = h_b = noise tile, in A-fragment layout.
  for (int i = threadIdx.x; i < 2048; i += blockDim.x) {
    const int c    = i >> 9;              // chunk 0..3 (fwd: 0,1  bwd: 2,3)
    const int lp   = (i >> 4) & 31;
    const int slot = i & 15;
    const int kh = lp >> 4, m = lp & 15;
    const int v = slot >> 1, j = slot & 1;
    const int k = kh * 8 + ((v >> 2) << 4) + ((v & 3) << 1) + j;
    hcat[i] = (_Float16)noise[(size_t)(rowbase + m) * HDIM + ((c & 1) << 5) + k];
  }
  for (int i = threadIdx.x; i < 1024; i += blockDim.x) xbuf[i] = (_Float16)0.0f;

  // h_prev in registers in C-fragment layout: (m = v + 8*(lane>>4), n = nloc).
  const int nloc = (w << 4) + (lane & 15);   // hidden column owned by this lane
  v8f hprev[2];
#pragma unroll
  for (int v = 0; v < 8; ++v) {
    const int m = v + ((lane >> 4) << 3);
    const float hv = noise[(size_t)(rowbase + m) * HDIM + nloc];
    hprev[0][v] = hv;
    hprev[1][v] = hv;
  }

  // Per-lane store mapping into fragment-linear state chunks.
  // Column within 32-wide chunk: k = ((w&1)<<4) + (lane&15).
  const int chunk_w = w >> 1;                                     // chunk within 64 cols
  const int slot_s  = ((w & 1) << 3) + (((lane >> 1) & 3) << 1) + (lane & 1);
  const int khs16   = ((lane >> 3) & 1) << 4;                     // kh<<4 for stores
  const int mbase   = (lane >> 4) << 3;                           // row base (0 or 8)

  // Per-lane biases (column-dependent only).
  float rb[2], zb[2], inb[2], hnb[2];
  rb[0]  = b_ih_f[nloc]       + b_hh_f[nloc];
  zb[0]  = b_ih_f[64 + nloc]  + b_hh_f[64 + nloc];
  inb[0] = b_ih_f[128 + nloc];
  hnb[0] = b_hh_f[128 + nloc];
  rb[1]  = b_ih_b[nloc]       + b_hh_b[nloc];
  zb[1]  = b_ih_b[64 + nloc]  + b_hh_b[64 + nloc];
  inb[1] = b_ih_b[128 + nloc];
  hnb[1] = b_hh_b[128 + nloc];
  const float latb = b_lat[nloc];
  const float outb = (w < 2) ? b_out[nloc] : 0.0f;

  __syncthreads();

  for (int s = 0; s < SEQ; ++s) {
    const _Float16* hcur = hcat + ((s & 1) << 11);        // 4 chunks
    _Float16*       hnxt = hcat + (((s + 1) & 1) << 11);
    const _Float16* xcur = xbuf + ((s & 1) << 10);        // 2 chunks
    _Float16*       xnxt = xbuf + (((s + 1) & 1) << 10);

    // x A-fragments are shared by both directions: load once.
    const v16h ax0 = load_frag(xcur);
    const v16h ax1 = load_frag(xcur + 512);

    // ---- Phase A: both GRU cells, this wave's 16 hidden columns per direction.
#pragma unroll
    for (int d = 0; d < 2; ++d) {
      const _Float16* wih = d ? wihb : wihf;
      const _Float16* whh = d ? whhb : whhf;
      const v16h ah0 = load_frag(hcur + ((d << 1) + 0) * 512);
      const v16h ah1 = load_frag(hcur + ((d << 1) + 1) * 512);

      v8f ar = {}, az = {}, ain = {}, ahn = {};
      // r gate: nt = w ; z gate: nt = 4+w ; n gate: nt = 8+w (KT = 2)
      ar  = wmma_f16(ax0, load_frag(wih + (((w      << 1) + 0) << 9)), ar);
      ar  = wmma_f16(ax1, load_frag(wih + (((w      << 1) + 1) << 9)), ar);
      ar  = wmma_f16(ah0, load_frag(whh + (((w      << 1) + 0) << 9)), ar);
      ar  = wmma_f16(ah1, load_frag(whh + (((w      << 1) + 1) << 9)), ar);
      az  = wmma_f16(ax0, load_frag(wih + ((((4 + w) << 1) + 0) << 9)), az);
      az  = wmma_f16(ax1, load_frag(wih + ((((4 + w) << 1) + 1) << 9)), az);
      az  = wmma_f16(ah0, load_frag(whh + ((((4 + w) << 1) + 0) << 9)), az);
      az  = wmma_f16(ah1, load_frag(whh + ((((4 + w) << 1) + 1) << 9)), az);
      ain = wmma_f16(ax0, load_frag(wih + ((((8 + w) << 1) + 0) << 9)), ain);
      ain = wmma_f16(ax1, load_frag(wih + ((((8 + w) << 1) + 1) << 9)), ain);
      ahn = wmma_f16(ah0, load_frag(whh + ((((8 + w) << 1) + 0) << 9)), ahn);
      ahn = wmma_f16(ah1, load_frag(whh + ((((8 + w) << 1) + 1) << 9)), ahn);

      _Float16* hdst = hnxt + ((d << 1) + chunk_w) * 512;
#pragma unroll
      for (int v = 0; v < 8; ++v) {
        const float r = sigm_f(ar[v] + rb[d]);
        const float z = sigm_f(az[v] + zb[d]);
        const float n = tanh_f(ain[v] + inb[d] + r * (ahn[v] + hnb[d]));
        const float h = (1.0f - z) * n + z * hprev[d][v];
        hprev[d][v] = h;
        const int lp = (v + mbase) + khs16;   // fragment lane' for row m
        hdst[lp * 16 + slot_s] = (_Float16)h;
      }
    }
    __syncthreads();

    // ---- Phase B: latent projection lat = leaky(concat(hf,hb) @ W_lat^T + b).
    {
      v8f al = {};
#pragma unroll
      for (int kt = 0; kt < 4; ++kt)
        al = wmma_f16(load_frag(hnxt + kt * 512),
                      load_frag(wlat + (((w << 2) + kt) << 9)), al);

      _Float16* xdst = xnxt + chunk_w * 512;
      _Float16* sdst = seqb + chunk_w * 512;
#pragma unroll
      for (int v = 0; v < 8; ++v) {
        const float l = leaky_f(al[v] + latb);   // next x
        const int lp = (v + mbase) + khs16;
        xdst[lp * 16 + slot_s] = (_Float16)l;
        sdst[lp * 16 + slot_s] = (_Float16)leaky_f(l);  // second LeakyReLU
      }
    }
    __syncthreads();

    // ---- Phase C: fused output projection, waves 0..1 own F=32 columns.
    if (w < 2) {
      v8f ao = {};
      ao = wmma_f16(load_frag(seqb),
                    load_frag(wout + (((w << 1) + 0) << 9)), ao);
      ao = wmma_f16(load_frag(seqb + 512),
                    load_frag(wout + (((w << 1) + 1) << 9)), ao);
#pragma unroll
      for (int v = 0; v < 8; ++v) {
        const int m = v + mbase;
        out[(size_t)(rowbase + m) * ((size_t)SEQ * FDIM) + (size_t)s * FDIM + nloc] =
            sigm_f(ao[v] + outb);
      }
    }
  }
}

// ---------------- launch ----------------

extern "C" void kernel_launch(void* const* d_in, const int* in_sizes, int n_in,
                              void* d_out, int out_size, void* d_ws, size_t ws_size,
                              hipStream_t stream) {
  (void)in_sizes; (void)n_in; (void)d_ws; (void)ws_size; (void)out_size;
  const float* noise  = (const float*)d_in[0];
  const float* W_ih_f = (const float*)d_in[1];
  const float* W_hh_f = (const float*)d_in[2];
  const float* b_ih_f = (const float*)d_in[3];
  const float* b_hh_f = (const float*)d_in[4];
  const float* W_ih_b = (const float*)d_in[5];
  const float* W_hh_b = (const float*)d_in[6];
  const float* b_ih_b = (const float*)d_in[7];
  const float* b_hh_b = (const float*)d_in[8];
  const float* W_lat  = (const float*)d_in[9];
  const float* b_lat  = (const float*)d_in[10];
  const float* W_out  = (const float*)d_in[11];
  const float* b_out  = (const float*)d_in[12];
  float* out = (float*)d_out;

  // LDS: (12288*4 + 8192 + 2048 + 4096 + 2048 + 1024) f16 = 66560 * 2 bytes.
  const size_t smem_bytes = 66560u * sizeof(_Float16);  // 133,120 B (~130 KB)

  gru_gan_scan_kernel<<<dim3(BATCH / 16), dim3(128), smem_bytes, stream>>>(
      noise, W_ih_f, W_hh_f, b_ih_f, b_hh_f,
      W_ih_b, W_hh_b, b_ih_b, b_hh_b,
      W_lat, b_lat, W_out, b_out, out);
}
